// MultiHeadSelfAttention_62526133895212
// MI455X (gfx1250) — matrix operand order repaired and checked by reading, not compiled, not run on hardware
//
#include <hip/hip_runtime.h>
#include <hip/hip_bf16.h>

// ---------------------------------------------------------------------------
// MultiHeadSelfAttention for MI455X (gfx1250), f16 WMMA with f32 accumulation.
// B=4, S=4096, D=512, H=8, DH=64. ~172 GFLOP, ~300MB HBM -> compute bound.
// Attention K/V tiles staged to LDS per-workgroup with async copies (8x L2
// traffic reduction); GEMMs use 32x64 wave tiles (~21 FLOP/B from cache).
// ---------------------------------------------------------------------------

#define BATCH 4
#define SLEN  4096
#define DEMB  512
#define NH    8
#define DHEAD 64
#define MROWS (BATCH * SLEN)   // 16384

typedef __attribute__((ext_vector_type(16))) _Float16 v16h;
typedef __attribute__((ext_vector_type(8)))  float    v8f;
typedef __attribute__((ext_vector_type(4)))  int      v4i;

// Fragment: 16 f16 per lane = 8 VGPRs. Filled with two 16-byte loads.
union HFrag { v16h h; uint4 q[2]; };

// ---------------------------------------------------------------------------
// gfx1250 async global->LDS copy (16B per lane) + asynccnt wait.
// Builtin signature (from clang diagnostic): (AS1 v4i*, AS3 v4i*, imm, imm).
// ---------------------------------------------------------------------------
__device__ inline void async_copy16(void* lds, const void* gptr) {
#if __has_builtin(__builtin_amdgcn_global_load_async_to_lds_b128)
  __builtin_amdgcn_global_load_async_to_lds_b128(
      (__attribute__((address_space(1))) v4i*)(void*)gptr,
      (__attribute__((address_space(3))) v4i*)lds, 0, 0);
#else
  uint32_t laddr = (uint32_t)(uintptr_t)(__attribute__((address_space(3))) void*)lds;
  asm volatile("global_load_async_to_lds_b128 %0, %1, off"
               :: "v"(laddr), "v"((unsigned long long)(uintptr_t)gptr)
               : "memory");
#endif
}

__device__ inline void wait_async0() {
#if __has_builtin(__builtin_amdgcn_s_wait_asynccnt)
  __builtin_amdgcn_s_wait_asynccnt(0);
#else
  asm volatile("s_wait_asynccnt 0x0" ::: "memory");
#endif
}

// ---------------------------------------------------------------------------
// fp32 -> fp16 conversion (vectorized by 4)
// ---------------------------------------------------------------------------
__global__ void f32_to_f16_kernel(const float* __restrict__ src,
                                  _Float16* __restrict__ dst, int n) {
  int i = (blockIdx.x * blockDim.x + threadIdx.x) * 4;
  if (i + 3 < n) {
    float4 v = *(const float4*)(src + i);
    union { _Float16 h[4]; uint2 u; } pk;
    pk.h[0] = (_Float16)v.x; pk.h[1] = (_Float16)v.y;
    pk.h[2] = (_Float16)v.z; pk.h[3] = (_Float16)v.w;
    *(uint2*)(dst + i) = pk.u;
  }
}

// ---------------------------------------------------------------------------
// QKV projection: qkv[m,e] = sum_k x[m,k]*w_in[e,k] + b_in[e]
//   M=16384, N=1536, K=512.  One 32(M) x 64(N) tile per wave.
// Epilogue scatters into attention layouts:
//   Q,K : [b,h,s,dh]  (dh contiguous)  -- Q pre-scaled by 1/sqrt(DH)=0.125
//   V   : [b,h,dh,s]  (s contiguous, i.e. V^T)
// ---------------------------------------------------------------------------
__global__ __launch_bounds__(256) void qkv_gemm_kernel(
    const _Float16* __restrict__ X,     // [M][512]   (K contiguous)
    const _Float16* __restrict__ W,     // [1536][512] (K contiguous)
    const float*    __restrict__ bias,  // [1536]
    _Float16* __restrict__ Qh, _Float16* __restrict__ Kh,
    _Float16* __restrict__ Vt) {
  const int lane = threadIdx.x & 31;
  const int wid  = threadIdx.x >> 5;
  const int lo = lane & 15, hi = lane >> 4;

  const int NT = (3 * DEMB) / 64;        // 24 n-tiles
  int tile  = blockIdx.x * 8 + wid;      // 12288 wave-tiles total
  int mtile = tile / NT, ntile = tile % NT;
  int m0 = mtile * 32, n0 = ntile * 64;

  v8f c0[4] = {}, c1[4] = {};
  const _Float16* arow0 = X + (size_t)(m0 + lo) * DEMB;
  const _Float16* arow1 = X + (size_t)(m0 + 16 + lo) * DEMB;

  for (int k0 = 0; k0 < DEMB; k0 += 32) {
    HFrag a0, a1;                        // A 16x32: K {kh*8..+7, kh*8+16..+23}
    a0.q[0] = *(const uint4*)(arow0 + k0 + hi * 8);
    a0.q[1] = *(const uint4*)(arow0 + k0 + hi * 8 + 16);
    a1.q[0] = *(const uint4*)(arow1 + k0 + hi * 8);
    a1.q[1] = *(const uint4*)(arow1 + k0 + hi * 8 + 16);
#pragma unroll
    for (int n = 0; n < 4; ++n) {        // B 32x16: col lo, K {hi*8..+7, hi*8+16..+23}
      const _Float16* brow = W + (size_t)(n0 + n * 16 + lo) * DEMB + k0 + hi * 8;
      HFrag b;
      b.q[0] = *(const uint4*)(brow);
      b.q[1] = *(const uint4*)(brow + 16);
      c0[n] = __builtin_amdgcn_wmma_f32_16x16x32_f16(
          false, a0.h, false, b.h, (short)0, c0[n], false, false);
      c1[n] = __builtin_amdgcn_wmma_f32_16x16x32_f16(
          false, a1.h, false, b.h, (short)0, c1[n], false, false);
    }
  }

#pragma unroll
  for (int n = 0; n < 4; ++n) {
    int e = n0 + n * 16 + lo;
    float bv = bias[e];
    int sec = e >> 9;            // 0=Q 1=K 2=V (16-wide subtile never crosses 512)
    int d  = e & 511;
    int h  = d >> 6;
    int dh = d & 63;
#pragma unroll
    for (int r = 0; r < 8; ++r) {
#pragma unroll
      for (int half = 0; half < 2; ++half) {
        int m = m0 + half * 16 + hi * 8 + r;   // C layout: VGPR r -> row hi*8+r
        float v = (half ? c1[n][r] : c0[n][r]) + bv;
        int b = m >> 12;
        int s = m & 4095;
        size_t bh = (size_t)(b * NH + h);
        if (sec == 0)
          Qh[(bh * SLEN + s) * DHEAD + dh] = (_Float16)(v * 0.125f);
        else if (sec == 1)
          Kh[(bh * SLEN + s) * DHEAD + dh] = (_Float16)v;
        else
          Vt[(bh * DHEAD + dh) * SLEN + s] = (_Float16)v;
      }
    }
  }
}

// ---------------------------------------------------------------------------
// Flash attention. Grid: (B*H) * (S/128) blocks, 8 waves/block.
// All 8 waves share one (b,h): K/V 32-key tiles are async-staged to LDS once
// per workgroup (double buffered). Each wave owns 16 q-rows.
// Output: attn[b,s,h,dh] f16 (= [M][512] K-contiguous for out-proj A).
// ---------------------------------------------------------------------------
__global__ __launch_bounds__(256) void attn_kernel(
    const _Float16* __restrict__ Qh, const _Float16* __restrict__ Kh,
    const _Float16* __restrict__ Vt, _Float16* __restrict__ At) {
  __shared__ _Float16 kbuf[2][32 * 64];    // [key][dh]   4KB per buffer
  __shared__ _Float16 vbuf[2][64 * 32];    // [dh][key]   4KB per buffer
  __shared__ _Float16 pbuf[8 * 16 * 32];   // per-wave P staging, 8KB

  const int tid  = threadIdx.x;
  const int lane = tid & 31;
  const int wid  = tid >> 5;
  const int lo = lane & 15, hi = lane >> 4;

  const int qblocks = SLEN / 128;              // 32
  const int bh   = blockIdx.x / qblocks;       // 0..31
  const int qblk = blockIdx.x % qblocks;
  const int qbase = qblk * 128 + wid * 16;

  const _Float16* Qb = Qh + (size_t)bh * SLEN * DHEAD;
  const _Float16* Kb = Kh + (size_t)bh * SLEN * DHEAD;
  const _Float16* Vb = Vt + (size_t)bh * DHEAD * SLEN;

  // Resident Q fragments (A operand), dh 0..31 and 32..63.
  HFrag aq0, aq1;
  {
    const _Float16* qrow = Qb + (size_t)(qbase + lo) * DHEAD + hi * 8;
    aq0.q[0] = *(const uint4*)(qrow);
    aq0.q[1] = *(const uint4*)(qrow + 16);
    aq1.q[0] = *(const uint4*)(qrow + 32);
    aq1.q[1] = *(const uint4*)(qrow + 48);
  }

  v8f o0 = {}, o1 = {}, o2 = {}, o3 = {};
  float mrow[8], lrow[8];
#pragma unroll
  for (int r = 0; r < 8; ++r) { mrow[r] = -1e30f; lrow[r] = 0.0f; }

  _Float16* pw = pbuf + wid * (16 * 32);

  // Per-thread staging addresses: each thread copies 16B of K and 16B of V.
  // K tile (4KB) is contiguous in global: Kb + kb*64 .. +2048 halves.
  // V tile: 64 rows of 32 keys (64B) with global row stride SLEN halves.
  const int koff     = tid * 16;                // byte offset in K tile
  const int v_dh     = tid >> 2;                // 0..63
  const int v_within = (tid & 3) * 16;          // byte offset in 64B row

  // Stage tile 0 into buffer 0.
  async_copy16((char*)kbuf[0] + koff, (const char*)(Kb) + koff);
  async_copy16((char*)vbuf[0] + tid * 16,
               (const char*)(Vb + (size_t)v_dh * SLEN) + v_within);

  const int niter = SLEN / 32;   // 128
  for (int it = 0; it < niter; ++it) {
    const int cur = it & 1;
    wait_async0();        // my share of tile `it` is in LDS
    __syncthreads();      // everyone's share landed; other buffer free

    if (it + 1 < niter) {
      const int kb1 = (it + 1) * 32;
      async_copy16((char*)kbuf[cur ^ 1] + koff,
                   (const char*)(Kb + (size_t)kb1 * DHEAD) + koff);
      async_copy16((char*)vbuf[cur ^ 1] + tid * 16,
                   (const char*)(Vb + (size_t)v_dh * SLEN + kb1) + v_within);
    }

    const _Float16* kl = kbuf[cur];
    const _Float16* vl = vbuf[cur];

    // ---- scores: two 16x16 tiles over 32 keys (4 WMMAs) ----
    v8f st[2];
#pragma unroll
    for (int t = 0; t < 2; ++t) {
      const _Float16* krow = kl + (t * 16 + lo) * DHEAD + hi * 8;
      HFrag kf0, kf1;                   // B op: col=key lo, K = dh {hi*8..+7, hi*8+16..+23}
      kf0.q[0] = *(const uint4*)(krow);
      kf0.q[1] = *(const uint4*)(krow + 16);
      kf1.q[0] = *(const uint4*)(krow + 32);
      kf1.q[1] = *(const uint4*)(krow + 48);
      v8f acc = {};
      acc = __builtin_amdgcn_wmma_f32_16x16x32_f16(
          false, aq0.h, false, kf0.h, (short)0, acc, false, false);
      acc = __builtin_amdgcn_wmma_f32_16x16x32_f16(
          false, aq1.h, false, kf1.h, (short)0, acc, false, false);
      st[t] = acc;
    }

    // ---- online softmax over the 32 new columns ----
    float pv0[8], pv1[8];
#pragma unroll
    for (int r = 0; r < 8; ++r) {
      float v = fmaxf(st[0][r], st[1][r]);       // row spread over 16 lanes
      v = fmaxf(v, __shfl_xor(v, 1));
      v = fmaxf(v, __shfl_xor(v, 2));
      v = fmaxf(v, __shfl_xor(v, 4));
      v = fmaxf(v, __shfl_xor(v, 8));
      float mn = fmaxf(mrow[r], v);
      float sc = __expf(mrow[r] - mn);
      mrow[r] = mn;
      float p0 = __expf(st[0][r] - mn);
      float p1 = __expf(st[1][r] - mn);
      float rs = p0 + p1;
      rs += __shfl_xor(rs, 1);
      rs += __shfl_xor(rs, 2);
      rs += __shfl_xor(rs, 4);
      rs += __shfl_xor(rs, 8);
      lrow[r] = lrow[r] * sc + rs;
      pv0[r] = p0; pv1[r] = p1;
      o0[r] *= sc; o1[r] *= sc; o2[r] *= sc; o3[r] *= sc;
    }

    // ---- P: C-layout -> row-major LDS -> A-fragment (same-wave DS in order) ----
    {
      _Float16* dst = pw + (hi * 8) * 32 + lo;   // row hi*8+r, col lo / lo+16
#pragma unroll
      for (int r = 0; r < 8; ++r) {
        dst[r * 32]      = (_Float16)pv0[r];
        dst[r * 32 + 16] = (_Float16)pv1[r];
      }
    }
    HFrag pf;
    {
      const _Float16* src = pw + lo * 32 + hi * 8;
      pf.q[0] = *(const uint4*)(src);
      pf.q[1] = *(const uint4*)(src + 16);
    }

    // ---- accumulate O += P x V (4 WMMAs, dh n-tiles of 16) ----
#pragma unroll
    for (int n = 0; n < 4; ++n) {
      const _Float16* vrow = vl + (n * 16 + lo) * 32 + hi * 8;
      HFrag vf;                          // B op: col=dh, K = key {hi*8..+7, hi*8+16..+23}
      vf.q[0] = *(const uint4*)(vrow);
      vf.q[1] = *(const uint4*)(vrow + 16);
      v8f acc = (n == 0) ? o0 : (n == 1) ? o1 : (n == 2) ? o2 : o3;
      acc = __builtin_amdgcn_wmma_f32_16x16x32_f16(
          false, pf.h, false, vf.h, (short)0, acc, false, false);
      if (n == 0) o0 = acc; else if (n == 1) o1 = acc;
      else if (n == 2) o2 = acc; else o3 = acc;
    }
  }

  // ---- normalize and store attn[b, s, h*64+dh] as f16 ----
  const int b = bh >> 3, h = bh & 7;
#pragma unroll
  for (int r = 0; r < 8; ++r) {
    float il = 1.0f / lrow[r];
    size_t row = ((size_t)b * SLEN + qbase + hi * 8 + r) * DEMB + h * DHEAD + lo;
    At[row]      = (_Float16)(o0[r] * il);
    At[row + 16] = (_Float16)(o1[r] * il);
    At[row + 32] = (_Float16)(o2[r] * il);
    At[row + 48] = (_Float16)(o3[r] * il);
  }
}

// ---------------------------------------------------------------------------
// Output projection: out[m,e] = sum_k attn[m,k]*w_out[e,k] + b_out[e]
//   M=16384, N=512, K=512. One 32x64 tile per wave. fp32 output.
// ---------------------------------------------------------------------------
__global__ __launch_bounds__(256) void out_gemm_kernel(
    const _Float16* __restrict__ A,     // [M][512] K-contiguous
    const _Float16* __restrict__ W,     // [512][512] K-contiguous
    const float*    __restrict__ bias,  // [512]
    float* __restrict__ out) {
  const int lane = threadIdx.x & 31;
  const int wid  = threadIdx.x >> 5;
  const int lo = lane & 15, hi = lane >> 4;

  const int NT = DEMB / 64;             // 8
  int tile  = blockIdx.x * 8 + wid;     // 4096 wave-tiles
  int mtile = tile / NT, ntile = tile % NT;
  int m0 = mtile * 32, n0 = ntile * 64;

  v8f c0[4] = {}, c1[4] = {};
  const _Float16* arow0 = A + (size_t)(m0 + lo) * DEMB;
  const _Float16* arow1 = A + (size_t)(m0 + 16 + lo) * DEMB;

  for (int k0 = 0; k0 < DEMB; k0 += 32) {
    HFrag a0, a1;
    a0.q[0] = *(const uint4*)(arow0 + k0 + hi * 8);
    a0.q[1] = *(const uint4*)(arow0 + k0 + hi * 8 + 16);
    a1.q[0] = *(const uint4*)(arow1 + k0 + hi * 8);
    a1.q[1] = *(const uint4*)(arow1 + k0 + hi * 8 + 16);
#pragma unroll
    for (int n = 0; n < 4; ++n) {
      const _Float16* brow = W + (size_t)(n0 + n * 16 + lo) * DEMB + k0 + hi * 8;
      HFrag b;
      b.q[0] = *(const uint4*)(brow);
      b.q[1] = *(const uint4*)(brow + 16);
      c0[n] = __builtin_amdgcn_wmma_f32_16x16x32_f16(
          false, a0.h, false, b.h, (short)0, c0[n], false, false);
      c1[n] = __builtin_amdgcn_wmma_f32_16x16x32_f16(
          false, a1.h, false, b.h, (short)0, c1[n], false, false);
    }
  }

#pragma unroll
  for (int n = 0; n < 4; ++n) {
    int e = n0 + n * 16 + lo;
    float bv = bias[e];
#pragma unroll
    for (int r = 0; r < 8; ++r) {
      int m = m0 + hi * 8 + r;
      out[(size_t)m * DEMB + e]        = c0[n][r] + bv;
      out[(size_t)(m + 16) * DEMB + e] = c1[n][r] + bv;
    }
  }
}

// ---------------------------------------------------------------------------
extern "C" void kernel_launch(void* const* d_in, const int* in_sizes, int n_in,
                              void* d_out, int out_size, void* d_ws, size_t ws_size,
                              hipStream_t stream) {
  (void)in_sizes; (void)n_in; (void)out_size; (void)ws_size;
  const float* x     = (const float*)d_in[0];
  const float* w_in  = (const float*)d_in[1];
  const float* b_in  = (const float*)d_in[2];
  const float* w_out = (const float*)d_in[3];
  const float* b_out = (const float*)d_in[4];
  float* out = (float*)d_out;

  // Workspace carve-up (~82 MB total, all 256B-aligned chunks).
  char* p = (char*)d_ws;
  _Float16* xh  = (_Float16*)p; p += (size_t)MROWS * DEMB * 2;               // 16 MB
  _Float16* wih = (_Float16*)p; p += (size_t)3 * DEMB * DEMB * 2;            // 1.5 MB
  _Float16* woh = (_Float16*)p; p += (size_t)DEMB * DEMB * 2;                // 0.5 MB
  _Float16* Qh  = (_Float16*)p; p += (size_t)BATCH * NH * SLEN * DHEAD * 2;  // 16 MB
  _Float16* Kh  = (_Float16*)p; p += (size_t)BATCH * NH * SLEN * DHEAD * 2;  // 16 MB
  _Float16* Vt  = (_Float16*)p; p += (size_t)BATCH * NH * SLEN * DHEAD * 2;  // 16 MB
  _Float16* At  = (_Float16*)p; p += (size_t)MROWS * DEMB * 2;               // 16 MB

  f32_to_f16_kernel<<<8192, 256, 0, stream>>>(x, xh, MROWS * DEMB);
  f32_to_f16_kernel<<<768,  256, 0, stream>>>(w_in, wih, 3 * DEMB * DEMB);
  f32_to_f16_kernel<<<256,  256, 0, stream>>>(w_out, woh, DEMB * DEMB);

  // QKV GEMM: 512 m-tiles x 24 n-tiles = 12288 wave-tiles / 8 per block
  qkv_gemm_kernel<<<1536, 256, 0, stream>>>(xh, wih, b_in, Qh, Kh, Vt);

  // Attention: (B*H) * (S/128) = 32 * 32 = 1024 blocks
  attn_kernel<<<1024, 256, 0, stream>>>(Qh, Kh, Vt, At);

  // Out projection: 512 m-tiles x 8 n-tiles = 4096 wave-tiles / 8 per block
  out_gemm_kernel<<<512, 256, 0, stream>>>(At, woh, b_out, out);
}
